// SpeakerEmbedderGeeArrYou_62405874811235
// MI455X (gfx1250) — matrix operand order repaired and checked by reading, not compiled, not run on hardware
//
#include <hip/hip_runtime.h>
#include <math.h>

typedef __bf16 v16bf __attribute__((ext_vector_type(16)));
typedef __bf16 v8bf  __attribute__((ext_vector_type(8)));
typedef float  v8f   __attribute__((ext_vector_type(8)));

#define N_MELS 80
#define KPAD   96           // 80 padded to 3*32 for WMMA K-loop
#define H      1024
#define G3     3072
#define BATCH  64
#define FRAMES 256
#define ROWS   (BATCH*FRAMES)   // 16384
#define NWG    32               // hidden-dim slices (persistent WGs)
#define SH     32               // hidden columns per WG
#define NLOC   96               // 3 gates * SH rows per WG

// ---------- scalar bf16 helpers (RNE, bit-exact, no cast dependence) ----------
__device__ __forceinline__ __bf16 f2bf(float f) {
  unsigned u = __float_as_uint(f);
  u = u + 0x7FFFu + ((u >> 16) & 1u);
  unsigned short hs = (unsigned short)(u >> 16);
  __bf16 r; __builtin_memcpy(&r, &hs, 2);
  return r;
}
__device__ __forceinline__ float bf2f(__bf16 b) {
  unsigned short hs; __builtin_memcpy(&hs, &b, 2);
  return __uint_as_float(((unsigned)hs) << 16);
}

// ---------- WMMA fragment loads (per ISA 16-bit A / B layouts) ----------
// A 16x32: lane m(0-15) holds row m K=[0..7]+[16..23]; lane m+16 holds K=[8..15]+[24..31]
__device__ __forceinline__ v16bf frag_a(const __bf16* tile, int ld, int lane) {
  const __bf16* p = tile + (long)(lane & 15) * ld + ((lane >> 4) << 3);
  v8bf lo = *(const v8bf*)p;
  v8bf hi = *(const v8bf*)(p + 16);
  return __builtin_shufflevector(lo, hi, 0,1,2,3,4,5,6,7,8,9,10,11,12,13,14,15);
}
// B 32x16 with B[k,n]=W[n,k]: lane n(0-15) holds K=[0..7]+[16..23] of row n;
// lane n+16 holds K=[8..15]+[24..31] of row n.
__device__ __forceinline__ v16bf frag_b(const __bf16* wtile, int ld, int lane) {
  const __bf16* p = wtile + (long)(lane & 15) * ld + ((lane >> 4) << 3);
  v8bf lo = *(const v8bf*)p;
  v8bf hi = *(const v8bf*)(p + 16);
  return __builtin_shufflevector(lo, hi, 0,1,2,3,4,5,6,7,8,9,10,11,12,13,14,15);
}
__device__ __forceinline__ v8f wmma_bf16(v16bf a, v16bf b, v8f c) {
  return __builtin_amdgcn_wmma_f32_16x16x32_bf16(false, a, false, b, (short)0, c, false, false);
}

// ---------- weight / input conversion ----------
__global__ void k_cvt(const float* __restrict__ s, __bf16* __restrict__ d, long n) {
  long i = (long)blockIdx.x * blockDim.x + threadIdx.x;
  long st = (long)gridDim.x * blockDim.x;
  for (; i < n; i += st) d[i] = f2bf(s[i]);
}
__global__ void k_pad(const float* __restrict__ s, __bf16* __restrict__ d,
                      int rows, int sc, int dc) {
  long n = (long)rows * dc;
  long i = (long)blockIdx.x * blockDim.x + threadIdx.x;
  long st = (long)gridDim.x * blockDim.x;
  for (; i < n; i += st) {
    int r = (int)(i / dc), c = (int)(i % dc);
    d[i] = (c < sc) ? f2bf(s[(long)r * sc + c]) : f2bf(0.0f);
  }
}

// ---------- time-parallel xp GEMM (1x4 register-blocked):
// xp[16384 x 3072] = A[16384 x K] @ W[3072 x K]^T + bih
__global__ __launch_bounds__(256) void k_xp(const __bf16* __restrict__ A,
                                            const __bf16* __restrict__ W,
                                            const float* __restrict__ bih,
                                            __bf16* __restrict__ xp, int K) {
  const int lane = threadIdx.x & 31, w = threadIdx.x >> 5;
  long grp = (long)blockIdx.x * 8 + w;          // tile-group = 16 rows x 64 cols
  int n_grp  = (int)(grp % (G3 / 64));          // 48 col-groups
  int m_tile = (int)(grp / (G3 / 64));
  if (m_tile >= ROWS / 16) return;              // uniform per wave: EXEC stays all-ones
  v8f acc[4];
#pragma unroll
  for (int j = 0; j < 4; ++j) acc[j] = (v8f){0.f,0.f,0.f,0.f,0.f,0.f,0.f,0.f};
  const __bf16* Ab = A + (long)m_tile * 16 * K;
  const __bf16* Wb = W + (long)n_grp * 64 * K;
  for (int k0 = 0; k0 < K; k0 += 32) {
    v16bf a = frag_a(Ab + k0, K, lane);         // one A frag reused 4x
#pragma unroll
    for (int j = 0; j < 4; ++j)
      acc[j] = wmma_bf16(a, frag_b(Wb + (long)j * 16 * K + k0, K, lane), acc[j]);
  }
  const int n = lane & 15;
  const int mb = m_tile * 16 + ((lane >> 4) << 3);
#pragma unroll
  for (int j = 0; j < 4; ++j) {
    int col = n_grp * 64 + j * 16 + n;
    float bias = bih[col];
#pragma unroll
    for (int r = 0; r < 8; ++r)
      xp[(long)(mb + r) * G3 + col] = f2bf(acc[j][r] + bias);
  }
}

// ---------- persistent GRU recurrence: 32 WGs, whh slice resident in LDS, grid-sync/step ----------
__global__ __launch_bounds__(256) void k_rec(const __bf16* __restrict__ whh,   // [3072][1024]
                                             const float* __restrict__ bhh,    // [3072]
                                             const __bf16* __restrict__ xp,    // [16384][3072]
                                             float* __restrict__ h32,          // [64][1024] (final only)
                                             __bf16* __restrict__ h16,         // [64][1024]
                                             __bf16* __restrict__ yout,        // [16384][1024]
                                             unsigned* __restrict__ syncc, int write_y) {
  extern __shared__ char smem[];
  __bf16* lw = (__bf16*)smem;                                   // [96][1024] bf16 = 192 KB
  float*  lg = (float*)(smem + (size_t)NLOC * H * 2);           // [64][96] f32 gate staging
  const int g = blockIdx.x, tid = threadIdx.x;
  const int lane = tid & 31, w = tid >> 5;

  // Stage this WG's whh slice (rows {r,z,n} x [g*32, g*32+32)) into LDS once.
  for (int i = tid; i < NLOC * (H / 8); i += 256) {
    int r = i >> 7, ch = i & 127;
    int grow = (r >> 5) * H + g * SH + (r & 31);
    ((v8bf*)lw)[(long)r * (H / 8) + ch] = ((const v8bf*)(whh + (long)grow * H))[ch];
  }

  // Per-thread persistent state: each (b,col) element is owned by exactly one
  // thread for the whole sequence -> keep fp32 hidden state in registers and
  // hoist the loop-invariant bhh biases out of the time loop.
  float hreg[8], br_[8], bz_[8], bn_[8];
#pragma unroll
  for (int e = 0; e < 8; ++e) {
    int idx = tid * 8 + e;
    int col = g * SH + (idx & 31);
    hreg[e] = 0.f;
    br_[e] = bhh[col];
    bz_[e] = bhh[H + col];
    bn_[e] = bhh[2 * H + col];
  }
  __syncthreads();

  const int m_tile = w & 3;           // 4 x 16 batch rows
  const int nb = (w >> 2) * 3;        // 3 of 6 n-tiles per wave

  for (int t = 0; t < FRAMES; ++t) {
    { // warm L2/L0 for this step's xp slice (global_prefetch_b8)
      int idx = tid * 8, b = idx >> 5, c = idx & 31;
      __builtin_prefetch(xp + ((long)b * FRAMES + t) * G3 + g * SH + c, 0, 1);
    }
    v8f a0 = {0.f,0.f,0.f,0.f,0.f,0.f,0.f,0.f};
    v8f a1 = {0.f,0.f,0.f,0.f,0.f,0.f,0.f,0.f};
    v8f a2 = {0.f,0.f,0.f,0.f,0.f,0.f,0.f,0.f};
    const __bf16* Ab = h16 + (long)m_tile * 16 * H;
    const __bf16* B0 = lw + (long)(nb + 0) * 16 * H;
    const __bf16* B1 = lw + (long)(nb + 1) * 16 * H;
    const __bf16* B2 = lw + (long)(nb + 2) * 16 * H;
    for (int k0 = 0; k0 < H; k0 += 32) {
      v16bf a = frag_a(Ab + k0, H, lane);
      a0 = wmma_bf16(a, frag_b(B0 + k0, H, lane), a0);
      a1 = wmma_bf16(a, frag_b(B1 + k0, H, lane), a1);
      a2 = wmma_bf16(a, frag_b(B2 + k0, H, lane), a2);
    }
    { // spill gate tiles to LDS staging [64][96] (f32 C/D layout)
      int n = lane & 15;
      int mb2 = m_tile * 16 + ((lane >> 4) << 3);
#pragma unroll
      for (int r = 0; r < 8; ++r) {
        lg[(mb2 + r) * NLOC + (nb + 0) * 16 + n] = a0[r];
        lg[(mb2 + r) * NLOC + (nb + 1) * 16 + n] = a1[r];
        lg[(mb2 + r) * NLOC + (nb + 2) * 16 + n] = a2[r];
      }
    }
    __syncthreads();
    // gate combine: 64x32 elements, 8 per thread; h state lives in hreg[]
#pragma unroll
    for (int e = 0; e < 8; ++e) {
      int idx = tid * 8 + e;
      int b = idx >> 5, c = idx & 31;
      int col = g * SH + c;
      long xrow = ((long)b * FRAMES + t) * G3;
      float rr = bf2f(xp[xrow + col])       + lg[b * NLOC + c]          + br_[e];
      float zz = bf2f(xp[xrow + H + col])   + lg[b * NLOC + SH + c]     + bz_[e];
      float xn = bf2f(xp[xrow + 2 * H + col]);
      float gn =                              lg[b * NLOC + 2 * SH + c] + bn_[e];
      float r_ = 1.f / (1.f + __expf(-rr));
      float z_ = 1.f / (1.f + __expf(-zz));
      float n_ = tanhf(xn + r_ * gn);
      float hn = (1.f - z_) * n_ + z_ * hreg[e];
      hreg[e] = hn;
      __bf16 hb = f2bf(hn);
      h16[(long)b * H + col] = hb;
      if (write_y) yout[((long)b * FRAMES + t) * H + col] = hb;
      if (t == FRAMES - 1) h32[(long)b * H + col] = hn;   // final state for projection
    }
    __syncthreads();
    // grid-wide step barrier: monotonic counter, release on arrive, acquire after
    if (tid == 0) {
      __threadfence();
      atomicAdd(syncc, 1u);
      unsigned tgt = (unsigned)NWG * (unsigned)(t + 1);
      while (__hip_atomic_load(syncc, __ATOMIC_ACQUIRE, __HIP_MEMORY_SCOPE_AGENT) < tgt)
        __builtin_amdgcn_s_sleep(1);
    }
    __syncthreads();
    __builtin_amdgcn_fence(__ATOMIC_ACQUIRE, "agent");
  }
}

// ---------- final projection + L2 normalize (fp32 for output accuracy) ----------
__global__ __launch_bounds__(256) void k_proj(const float* __restrict__ h,
                                              const float* __restrict__ wp,
                                              const float* __restrict__ bp,
                                              float* __restrict__ out) {
  __shared__ float red[256];
  int b = blockIdx.x, c = threadIdx.x;
  const float* hr = h + (long)b * H;
  const float* wr = wp + (long)c * H;
  float acc = bp[c];
#pragma unroll 4
  for (int k = 0; k < H; ++k) acc = fmaf(hr[k], wr[k], acc);
  red[c] = acc * acc;
  __syncthreads();
  for (int s = 128; s > 0; s >>= 1) {
    if (c < s) red[c] += red[c + s];
    __syncthreads();
  }
  float inv = 1.f / fmaxf(sqrtf(red[0]), 1e-12f);
  out[(long)b * 256 + c] = acc * inv;
}

extern "C" void kernel_launch(void* const* d_in, const int* in_sizes, int n_in,
                              void* d_out, int out_size, void* d_ws, size_t ws_size,
                              hipStream_t stream) {
  (void)in_sizes; (void)n_in; (void)out_size; (void)ws_size;
  const float* x    = (const float*)d_in[0];
  const float* wih0 = (const float*)d_in[1];
  const float* whh0 = (const float*)d_in[2];
  const float* bih0 = (const float*)d_in[3];
  const float* bhh0 = (const float*)d_in[4];
  const float* wihr = (const float*)d_in[5];
  const float* whhr = (const float*)d_in[6];
  const float* bihr = (const float*)d_in[7];
  const float* bhhr = (const float*)d_in[8];
  const float* wp   = (const float*)d_in[9];
  const float* bp   = (const float*)d_in[10];

  char* ws = (char*)d_ws;                       // offsets all 256B-aligned
  __bf16*   whhB  = (__bf16*)(ws + 0);          // 3*3072*1024 bf16   (18,874,368)
  __bf16*   wih0p = (__bf16*)(ws + 18874368);   // 3072*96            (589,824)
  __bf16*   wihrB = (__bf16*)(ws + 19464192);   // 2*3072*1024        (12,582,912)
  __bf16*   x0p   = (__bf16*)(ws + 32047104);   // 16384*96           (3,145,728)
  __bf16*   yA    = (__bf16*)(ws + 35192832);   // 16384*1024         (33,554,432)
  __bf16*   yB    = (__bf16*)(ws + 68747264);   // 16384*1024
  __bf16*   xpB   = (__bf16*)(ws + 102301696);  // 16384*3072         (100,663,296)
  float*    h32   = (float*) (ws + 202964992);  // 64*1024 f32
  __bf16*   h16   = (__bf16*)(ws + 203227136);  // 64*1024 bf16
  unsigned* syncc = (unsigned*)(ws + 203358208);

  // precision conversion / padding (one-shot, parallel)
  k_cvt<<<2048, 256, 0, stream>>>(whh0, whhB, (long)G3 * H);
  k_cvt<<<4096, 256, 0, stream>>>(whhr, whhB + (long)G3 * H, 2L * G3 * H);
  k_cvt<<<4096, 256, 0, stream>>>(wihr, wihrB, 2L * G3 * H);
  k_pad<<<1024, 256, 0, stream>>>(wih0, wih0p, G3, N_MELS, KPAD);
  k_pad<<<2048, 256, 0, stream>>>(x, x0p, ROWS, N_MELS, KPAD);

  const size_t smem = (size_t)NLOC * H * 2 + (size_t)BATCH * NLOC * 4;  // 221,184 B
  for (int l = 0; l < 3; ++l) {
    const __bf16* A   = (l == 0) ? x0p : (l == 1 ? yA : yB);
    const int     K   = (l == 0) ? KPAD : H;
    const __bf16* W   = (l == 0) ? wih0p : wihrB + (long)(l - 1) * G3 * H;
    const float*  bih = (l == 0) ? bih0 : bihr + (l - 1) * G3;
    k_xp<<<(ROWS / 16) * (G3 / 64) / 8, 256, 0, stream>>>(A, W, bih, xpB, K);

    hipMemsetAsync(h32, 0, (size_t)BATCH * H * 4, stream);
    hipMemsetAsync(h16, 0, (size_t)BATCH * H * 2, stream);
    hipMemsetAsync(syncc, 0, 256, stream);

    const __bf16* whhl = whhB + (long)l * G3 * H;
    const float*  bhhl = (l == 0) ? bhh0 : bhhr + (l - 1) * G3;
    __bf16*       yout = (l == 0) ? yA : yB;     // unused when l==2 (write_y=0)
    k_rec<<<NWG, 256, smem, stream>>>(whhl, bhhl, xpB, h32, h16, yout, syncc, (l < 2) ? 1 : 0);
  }
  k_proj<<<BATCH, 256, 0, stream>>>(h32, wp, bp, (float*)d_out);
}
